// MultiQueryAttention_28003186770597
// MI455X (gfx1250) — matrix operand order repaired and checked by reading, not compiled, not run on hardware
//
#include <hip/hip_runtime.h>

// ---------------------------------------------------------------------------
// MQA forward for MI455X (gfx1250): bf16 WMMA + async-LDS staging.
//   x:(2,2048,1024) qkv_w:(1152,1024) qkv_b:(1152) out_w:(1024,1024) out_b:(1024)
//   out:(2,2048,1024) f32
// ---------------------------------------------------------------------------

typedef unsigned short u16;
typedef __attribute__((ext_vector_type(16))) __bf16 v16bf;
typedef __attribute__((ext_vector_type(8)))  float  v8f;
typedef __attribute__((ext_vector_type(8)))  unsigned short v8us;

union Frag { v16bf bf; v8us h[2]; };
union Acc  { v8f v; float f[8]; };

#define WMMA_BF16(A_, B_, C_) \
  __builtin_amdgcn_wmma_f32_16x16x32_bf16(false, (A_), false, (B_), (short)0, (C_), false, false)

static __device__ __forceinline__ u16 f2bf(float f) {
  unsigned int u = __float_as_uint(f);
  u = (u + 0x7FFFu + ((u >> 16) & 1u)) >> 16;   // round-to-nearest-even
  return (u16)u;
}

// async 16B copy global -> LDS (tracked by ASYNCcnt)
static __device__ __forceinline__ void async_b128(unsigned lds_off,
                                                  unsigned long long gaddr) {
  asm volatile("global_load_async_to_lds_b128 %0, %1, off"
               :: "v"(lds_off), "v"(gaddr) : "memory");
}

// ---------------------------------------------------------------------------
// f32 -> bf16 bulk convert (grid-stride)
// ---------------------------------------------------------------------------
__global__ void k_f32_to_bf16(const float* __restrict__ in, u16* __restrict__ out, int n) {
  int i = blockIdx.x * blockDim.x + threadIdx.x;
  int stride = gridDim.x * blockDim.x;
  for (; i < n; i += stride) out[i] = f2bf(in[i]);
}

// ---------------------------------------------------------------------------
// C[M,N] = A[M,K](bf16) * W[N,K]^T(bf16) + bias
// block = 256 (8 waves); wave owns 32(M) x 64(N) -> 8 WMMA / k-step,
// each B fragment reused by two A fragments. Block owns 256 x 64.
// ---------------------------------------------------------------------------
__global__ __launch_bounds__(256) void k_gemm_bf16(
    const u16* __restrict__ A, const u16* __restrict__ W,
    const float* __restrict__ bias, float* __restrict__ C,
    int M, int N, int K) {
  const int wave = threadIdx.x >> 5;
  const int lane = threadIdx.x & 31;
  const int hi   = lane >> 4;     // which 16-lane half
  const int l16  = lane & 15;
  const int m0   = blockIdx.y * 256 + wave * 32;
  const int n0   = blockIdx.x * 64;

  const u16* arow0 = A + (size_t)(m0 + l16) * K + hi * 8;        // A-frag rows m0..m0+15
  const u16* arow1 = arow0 + (size_t)16 * K;                     // rows m0+16..m0+31
  const u16* brow  = W + (size_t)(n0 + l16) * K + hi * 8;        // B-frag (j adds 16*K)

  Acc acc[2][4];
  v8f z8 = {0.f, 0.f, 0.f, 0.f, 0.f, 0.f, 0.f, 0.f};
#pragma unroll
  for (int i = 0; i < 2; ++i)
#pragma unroll
    for (int j = 0; j < 4; ++j) acc[i][j].v = z8;

  for (int k0 = 0; k0 < K; k0 += 32) {
    Frag a0, a1;
    a0.h[0] = *(const v8us*)(arow0 + k0);
    a0.h[1] = *(const v8us*)(arow0 + k0 + 16);
    a1.h[0] = *(const v8us*)(arow1 + k0);
    a1.h[1] = *(const v8us*)(arow1 + k0 + 16);
    __builtin_prefetch(arow0 + k0 + 128, 0, 1);                  // global_prefetch_b8
    __builtin_prefetch(arow1 + k0 + 128, 0, 1);
#pragma unroll
    for (int j = 0; j < 4; ++j) {
      const u16* bp = brow + (size_t)j * 16 * K + k0;
      Frag b;
      b.h[0] = *(const v8us*)(bp);
      b.h[1] = *(const v8us*)(bp + 16);
      acc[0][j].v = WMMA_BF16(a0.bf, b.bf, acc[0][j].v);
      acc[1][j].v = WMMA_BF16(a1.bf, b.bf, acc[1][j].v);
    }
  }

#pragma unroll
  for (int j = 0; j < 4; ++j) {
    const int col = n0 + j * 16 + l16;
    const float bs = bias[col];
#pragma unroll
    for (int i = 0; i < 2; ++i)
#pragma unroll
      for (int r = 0; r < 8; ++r) {
        const int row = m0 + i * 16 + r + 8 * hi;
        C[(size_t)row * N + col] = acc[i][j].f[r] + bs;
      }
  }
}

// ---------------------------------------------------------------------------
// bias + RoPE + scatter into attention layouts:
//   qb: bf16 [B,H,S,64]  kb: bf16 [B,S,64]  vt: bf16 [B,64,S] (transposed!)
// one thread = one (even,odd) pair; 576 pairs per (b,s)
// ---------------------------------------------------------------------------
__global__ void k_rope_scatter(const float* __restrict__ qkv,
                               const float* __restrict__ qkv_b,
                               u16* __restrict__ qb, u16* __restrict__ kb,
                               u16* __restrict__ vt) {
  const int PP = 576;                       // 512 q-pairs + 32 k-pairs + 32 v-pairs
  int tid = blockIdx.x * blockDim.x + threadIdx.x;
  if (tid >= 2 * 2048 * PP) return;
  const int p  = tid % PP;
  const int bs = tid / PP;
  const int s  = bs % 2048;
  const int b  = bs / 2048;
  const float* row = qkv + (size_t)bs * 1152;

  if (p < 544) {                            // rope regions (q or k)
    int i, off;
    if (p < 512) { i = p & 31; off = (p >> 5) * 64 + 2 * i; }
    else         { i = p - 512; off = 1024 + 2 * i; }
    const float x1 = row[off]     + qkv_b[off];
    const float x2 = row[off + 1] + qkv_b[off + 1];
    // inv_freq = 10000^(-2i/64)
    const float inv = __expf(-(2.0f * (float)i / 64.0f) * 9.210340371976184f);
    float sn, cs;
    __sincosf((float)s * inv, &sn, &cs);
    const float o1 = x1 * cs - x2 * sn;
    const float o2 = x1 * sn + x2 * cs;
    if (p < 512) {
      const int head = p >> 5;
      const size_t idx = (((size_t)(b * 16 + head)) * 2048 + s) * 64 + 2 * i;
      qb[idx] = f2bf(o1); qb[idx + 1] = f2bf(o2);
    } else {
      const size_t idx = ((size_t)b * 2048 + s) * 64 + 2 * i;
      kb[idx] = f2bf(o1); kb[idx + 1] = f2bf(o2);
    }
  } else {                                  // v: no rope, store transposed
    const int d = 2 * (p - 544);
    const float v0 = row[1088 + d]     + qkv_b[1088 + d];
    const float v1 = row[1088 + d + 1] + qkv_b[1088 + d + 1];
    const size_t i0 = ((size_t)b * 64 + d) * 2048 + s;
    vt[i0]        = f2bf(v0);
    vt[i0 + 2048] = f2bf(v1);
  }
}

// ---------------------------------------------------------------------------
// causal flash attention, 1 wave per (b, h, 16-query tile)
//   K/V tiles double-buffered in LDS via global_load_async_to_lds_b128
//   scores = q@k^T / 8  -> online softmax -> o += P@v
// ---------------------------------------------------------------------------
__global__ __launch_bounds__(32) void k_flash(const u16* __restrict__ qb,
                                              const u16* __restrict__ kb,
                                              const u16* __restrict__ vt,
                                              float* __restrict__ ctx) {
  const int q0 = blockIdx.x * 16;
  const int h  = blockIdx.y;
  const int b  = blockIdx.z;
  const int lane = threadIdx.x & 31;
  const int hi   = lane >> 4;
  const int l16  = lane & 15;

  __shared__ alignas(16) u16 lk[2][32 * 64];   // K tiles: [key rel][dim], 2 x 4KB
  __shared__ alignas(16) u16 lv[2][64 * 32];   // V tiles: [dim][key rel], 2 x 4KB
  __shared__ alignas(16) u16 pt[16 * 32];      // P tile (C-layout -> A-layout)

  const unsigned lk_off[2] = { (unsigned)(unsigned long long)&lk[0][0],
                               (unsigned)(unsigned long long)&lk[1][0] };
  const unsigned lv_off[2] = { (unsigned)(unsigned long long)&lv[0][0],
                               (unsigned)(unsigned long long)&lv[1][0] };
  const char* kB = (const char*)(kb + (size_t)b * 2048 * 64);   // key row = 128 B
  const char* vB = (const char*)(vt + (size_t)b * 64 * 2048);   // dim row = 4096 B

  // stage one 32-key tile (K: 4KB contiguous; V: 64 rows x 64B) -> 16 async ops
  auto stage = [&](int buf, int kt) {
#pragma unroll
    for (int c = 0; c < 8; ++c)   // K tile: contiguous 4096 B
      async_b128(lk_off[buf] + c * 512 + lane * 16,
                 (unsigned long long)(kB + (size_t)kt * 128 + c * 512 + lane * 16));
#pragma unroll
    for (int c = 0; c < 8; ++c) { // V tile: 64 rows of 64 B (stride 4096 B)
      const int vrow = c * 8 + (lane >> 2);
      async_b128(lv_off[buf] + vrow * 64 + (lane & 3) * 16,
                 (unsigned long long)(vB + (size_t)vrow * 4096 + (size_t)kt * 2 +
                                      (lane & 3) * 16));
    }
  };

  // Q fragments (reused across all key tiles): D=64 -> two K=32 fragments
  const u16* qrow = qb + (((size_t)(b * 16 + h)) * 2048 + (q0 + l16)) * 64 + hi * 16;
  Frag qf0, qf1;
  qf0.h[0] = *(const v8us*)(qrow);      qf0.h[1] = *(const v8us*)(qrow + 8);
  qf1.h[0] = *(const v8us*)(qrow + 32); qf1.h[1] = *(const v8us*)(qrow + 40);

  v8f z8 = {0.f, 0.f, 0.f, 0.f, 0.f, 0.f, 0.f, 0.f};
  Acc o[4];
#pragma unroll
  for (int j = 0; j < 4; ++j) o[j].v = z8;
  float mi[8], Li[8];
#pragma unroll
  for (int r = 0; r < 8; ++r) { mi[r] = -1e30f; Li[r] = 0.f; }

  const int nt = ((q0 + 15) >> 5) + 1;      // tiles touching causal region
  stage(0, 0);
  for (int t = 0; t < nt; ++t) {
    const int kt  = t * 32;
    const int buf = t & 1;
    if (t + 1 < nt) {
      stage((t + 1) & 1, kt + 32);
      asm volatile("s_wait_asynccnt 0x10" ::: "memory");  // tile t landed in LDS
    } else {
      asm volatile("s_wait_asynccnt 0x0" ::: "memory");
    }

    // ---- scores from LDS: 2 N-tiles x 2 K-steps = 4 WMMA ----
    Acc s0, s1;
    {
      const u16* kr0 = &lk[buf][(l16) * 64 + hi * 16];        // keys kt..kt+15
      const u16* kr1 = &lk[buf][(16 + l16) * 64 + hi * 16];   // keys kt+16..kt+31
      Frag k00, k01, k10, k11;
      k00.h[0] = *(const v8us*)(kr0);      k00.h[1] = *(const v8us*)(kr0 + 8);
      k01.h[0] = *(const v8us*)(kr0 + 32); k01.h[1] = *(const v8us*)(kr0 + 40);
      k10.h[0] = *(const v8us*)(kr1);      k10.h[1] = *(const v8us*)(kr1 + 8);
      k11.h[0] = *(const v8us*)(kr1 + 32); k11.h[1] = *(const v8us*)(kr1 + 40);
      s0.v = WMMA_BF16(qf0.bf, k00.bf, z8);
      s0.v = WMMA_BF16(qf1.bf, k01.bf, s0.v);
      s1.v = WMMA_BF16(qf0.bf, k10.bf, z8);
      s1.v = WMMA_BF16(qf1.bf, k11.bf, s1.v);
    }

    // ---- scale + causal mask + online softmax ----
#pragma unroll
    for (int r = 0; r < 8; ++r) {
      const int row  = q0 + r + 8 * hi;
      const int key0 = kt + l16;
      const int key1 = kt + 16 + l16;
      float a = s0.f[r] * 0.125f;
      float c = s1.f[r] * 0.125f;
      if (key0 > row) a = -1e30f;
      if (key1 > row) c = -1e30f;
      float mx = fmaxf(a, c);
      mx = fmaxf(mx, __shfl_xor(mx, 1, 32));
      mx = fmaxf(mx, __shfl_xor(mx, 2, 32));
      mx = fmaxf(mx, __shfl_xor(mx, 4, 32));
      mx = fmaxf(mx, __shfl_xor(mx, 8, 32));
      const float mn = fmaxf(mi[r], mx);
      const float p0 = __expf(a - mn);
      const float p1 = __expf(c - mn);
      float rs = p0 + p1;
      rs += __shfl_xor(rs, 1, 32);
      rs += __shfl_xor(rs, 2, 32);
      rs += __shfl_xor(rs, 4, 32);
      rs += __shfl_xor(rs, 8, 32);
      const float resc = __expf(mi[r] - mn);
      Li[r] = Li[r] * resc + rs;
      mi[r] = mn;
#pragma unroll
      for (int j = 0; j < 4; ++j) o[j].f[r] *= resc;
      pt[(r + 8 * hi) * 32 + l16]      = f2bf(p0);   // C-layout -> LDS
      pt[(r + 8 * hi) * 32 + 16 + l16] = f2bf(p1);
    }

    asm volatile("s_wait_dscnt 0" ::: "memory");     // P stores visible before reload

    // ---- reload P as A-fragment (16x32, K = keys) ----
    Frag pf;
    pf.h[0] = *(const v8us*)(&pt[l16 * 32 + hi * 8]);
    pf.h[1] = *(const v8us*)(&pt[l16 * 32 + hi * 8 + 16]);

    // ---- o += P @ V from LDS: 4 N-tiles over D=64, K=32 each ----
#pragma unroll
    for (int j = 0; j < 4; ++j) {
      const u16* vr = &lv[buf][(j * 16 + l16) * 32 + hi * 8];
      Frag vf;
      vf.h[0] = *(const v8us*)(vr);
      vf.h[1] = *(const v8us*)(vr + 16);
      o[j].v = WMMA_BF16(pf.bf, vf.bf, o[j].v);
    }
  }

  // ---- normalize + write ctx [B,S,E] head-major ----
#pragma unroll
  for (int j = 0; j < 4; ++j) {
    const int col = h * 64 + j * 16 + l16;
#pragma unroll
    for (int r = 0; r < 8; ++r) {
      const int row = q0 + r + 8 * hi;
      ctx[((size_t)b * 2048 + row) * 1024 + col] = o[j].f[r] / Li[r];
    }
  }
}

// ---------------------------------------------------------------------------
// host launcher
// ---------------------------------------------------------------------------
extern "C" void kernel_launch(void* const* d_in, const int* in_sizes, int n_in,
                              void* d_out, int out_size, void* d_ws, size_t ws_size,
                              hipStream_t stream) {
  const float* x      = (const float*)d_in[0];   // [2,2048,1024]
  const float* qkv_w  = (const float*)d_in[1];   // [1152,1024]
  const float* qkv_b  = (const float*)d_in[2];   // [1152]
  const float* out_w  = (const float*)d_in[3];   // [1024,1024]
  const float* out_b  = (const float*)d_in[4];   // [1024]
  float* out = (float*)d_out;                    // [2,2048,1024]

  char* ws = (char*)d_ws;
  u16*   xb   = (u16*)(ws + 0);                  //  8,388,608 B  x bf16
  u16*   qwb  = (u16*)(ws + 8388608);            //  2,359,296 B  qkv_w bf16
  u16*   owb  = (u16*)(ws + 10747904);           //  2,097,152 B  out_w bf16
  float* qkv  = (float*)(ws + 12845056);         // 18,874,368 B  qkv f32
  u16*   qb   = (u16*)(ws + 31719424);           //  8,388,608 B  q roped bf16 [B,H,S,D]
  u16*   kb   = (u16*)(ws + 40108032);           //    524,288 B  k roped bf16 [B,S,D]
  u16*   vt   = (u16*)(ws + 40632320);           //    524,288 B  v bf16 [B,D,S]
  float* ctx  = (float*)(ws + 41156608);         // 16,777,216 B  attn out f32
  u16*   ctxb = (u16*)(ws + 57933824);           //  8,388,608 B  attn out bf16

  k_f32_to_bf16<<<8192, 256, 0, stream>>>(x, xb, 2 * 2048 * 1024);
  k_f32_to_bf16<<<4608, 256, 0, stream>>>(qkv_w, qwb, 1152 * 1024);
  k_f32_to_bf16<<<4096, 256, 0, stream>>>(out_w, owb, 1024 * 1024);

  // qkv = x @ qkv_w^T + b : M=4096 N=1152 K=1024
  k_gemm_bf16<<<dim3(1152 / 64, 4096 / 256), 256, 0, stream>>>(
      xb, qwb, qkv_b, qkv, 4096, 1152, 1024);

  k_rope_scatter<<<(2 * 2048 * 576) / 256, 256, 0, stream>>>(qkv, qkv_b, qb, kb, vt);

  k_flash<<<dim3(2048 / 16, 16, 2), 32, 0, stream>>>(qb, kb, vt, ctx);

  k_f32_to_bf16<<<8192, 256, 0, stream>>>(ctx, ctxb, 2 * 2048 * 1024);

  // out = ctx @ out_w^T + b : M=4096 N=1024 K=1024
  k_gemm_bf16<<<dim3(1024 / 64, 4096 / 256), 256, 0, stream>>>(
      ctxb, owb, out_b, out, 4096, 1024, 1024);
}
